// MultiHeadAttention_53326313947600
// MI455X (gfx1250) — matrix operand order repaired and checked by reading, not compiled, not run on hardware
//
#include <hip/hip_runtime.h>

// MHA forward for B=16,S=512,D=768,H=12,DH=64 on gfx1250 (wave32, WMMA bf16).
// d_out = [out (16*512*768 f32)] ++ [attn (16*12*512*512 f32)]
// d_ws  = Qh,Kh (B,H,S,DH) ++ VhT (B,H,DH,S) ++ ctx (B,S,D) ++ 4x W^T, all bf16 (~55 MB)

typedef __bf16 bf16;
typedef __attribute__((ext_vector_type(4)))  bf16  bf16x4;
typedef __attribute__((ext_vector_type(8)))  bf16  bf16x8;
typedef __attribute__((ext_vector_type(16))) bf16  bf16x16;
typedef __attribute__((ext_vector_type(8)))  float floatx8;
typedef __attribute__((ext_vector_type(4)))  int   i32x4;

#define AS1 __attribute__((address_space(1)))
#define AS3 __attribute__((address_space(3)))

#define BATCH 16
#define SEQ   512
#define DIM   768
#define NH    12
#define DHD   64

#ifndef __has_builtin
#define __has_builtin(x) 0
#endif
#if __has_builtin(__builtin_amdgcn_global_load_async_to_lds_b128)
#define HAVE_ASYNC_LDS 1
#endif

// 16-byte global -> LDS copy. Async (ASYNCcnt) when the gfx1250 builtin exists,
// else a synchronous VGPR bounce (correct under the same __syncthreads fence).
__device__ __forceinline__ void async_cp16(const bf16* __restrict__ g, bf16* l) {
#ifdef HAVE_ASYNC_LDS
  __builtin_amdgcn_global_load_async_to_lds_b128(
      (AS1 i32x4*)g, (AS3 i32x4*)l, 0, 0);
#else
  *(bf16x8*)l = *(const bf16x8*)g;
#endif
}

__device__ __forceinline__ void wait_async() {
#ifdef HAVE_ASYNC_LDS
#if __has_builtin(__builtin_amdgcn_s_wait_asynccnt)
  __builtin_amdgcn_s_wait_asynccnt(0);
#else
  asm volatile("s_wait_asynccnt 0x0" ::: "memory");
#endif
#endif
}

union Frag16 { bf16x16 f; bf16x8 h[2]; };

__device__ __forceinline__ bf16x16 load_frag(const bf16* p0, const bf16* p1) {
  Frag16 fr;
  fr.h[0] = *(const bf16x8*)p0;   // 16B chunk
  fr.h[1] = *(const bf16x8*)p1;   // 16B chunk
  return fr.f;
}

__device__ __forceinline__ floatx8 wmma_bf16(bf16x16 a, bf16x16 b, floatx8 c) {
  // D = A(16x32) * B(32x16) + C, fp32 accumulate
  return __builtin_amdgcn_wmma_f32_16x16x32_bf16(false, a, false, b, (short)0, c,
                                                 false, false);
}

__device__ __forceinline__ bf16x4 cvt4(float4 v) {
  bf16x4 t = { (bf16)v.x, (bf16)v.y, (bf16)v.z, (bf16)v.w };
  return t;
}

// ---------------------------------------------------------------------------
// Kernel 0: W -> W^T bf16 (once per weight). grid=(144,4), block=256.
// WT[mat][n][k] = W[k][n]; makes GEMM B-tiles contiguous (async-LDS eligible).
// ---------------------------------------------------------------------------
__global__ void __launch_bounds__(256) prep_wt_kernel(
    const float* __restrict__ Wq, const float* __restrict__ Wk,
    const float* __restrict__ Wv, const float* __restrict__ Wo,
    bf16* __restrict__ WT)
{
  const int mat = blockIdx.y;
  const float* W = (mat == 0) ? Wq : (mat == 1) ? Wk : (mat == 2) ? Wv : Wo;
  bf16* dst = WT + (size_t)mat * DIM * DIM;
  const int tk = (blockIdx.x / (DIM / 64)) * 64;   // k block
  const int tn = (blockIdx.x % (DIM / 64)) * 64;   // n block

  __shared__ float tile[64][65];                   // +1 pad: conflict-free columns
  const int tid = threadIdx.x;
  const int r = tid >> 4, c4 = (tid & 15) * 4;

  #pragma unroll
  for (int i = 0; i < 4; ++i) {
    const int rr = r + 16 * i;
    float4 x = *(const float4*)&W[(size_t)(tk + rr) * DIM + tn + c4];
    tile[rr][c4 + 0] = x.x; tile[rr][c4 + 1] = x.y;
    tile[rr][c4 + 2] = x.z; tile[rr][c4 + 3] = x.w;
  }
  __syncthreads();
  #pragma unroll
  for (int i = 0; i < 4; ++i) {
    const int rr = r + 16 * i;                     // n within tile
    bf16x4 t = { (bf16)tile[c4 + 0][rr], (bf16)tile[c4 + 1][rr],
                 (bf16)tile[c4 + 2][rr], (bf16)tile[c4 + 3][rr] };
    *(bf16x4*)&dst[(size_t)(tn + rr) * DIM + tk + c4] = t;
  }
}

// ---------------------------------------------------------------------------
// Kernel 1: Q/K/V projections. grid=(768,3), block=256 (8 waves).
// Double-buffered LDS, one barrier/iter. B-tile: async global->LDS b128 from
// W^T bf16. A-tile: fp32 float4 loads + cvt (needs conversion anyway).
// gemm 0/1 -> (B,H,S,DH) bf16 ; gemm 2 -> (B,H,DH,S) bf16 (transposed V).
// ---------------------------------------------------------------------------
__global__ void __launch_bounds__(256) qkv_proj_kernel(
    const float* __restrict__ q, const float* __restrict__ k, const float* __restrict__ v,
    const bf16* __restrict__ WT,
    const float* __restrict__ bq, const float* __restrict__ bk, const float* __restrict__ bv,
    bf16* __restrict__ Qh, bf16* __restrict__ Kh, bf16* __restrict__ VhT)
{
  const int gemm = blockIdx.y;
  const float* src  = (gemm == 0) ? q  : (gemm == 1) ? k  : v;
  const bf16*  Wt   = WT + (size_t)gemm * DIM * DIM;
  const float* bias = (gemm == 0) ? bq : (gemm == 1) ? bk : bv;
  bf16* dst         = (gemm == 0) ? Qh : (gemm == 1) ? Kh : VhT;

  const int tile = blockIdx.x;
  const int m0 = (tile / (DIM / 64)) * 128;
  const int n0 = (tile % (DIM / 64)) * 64;

  __shared__ bf16 Asl[2][128][32];   // 16 KB double-buffered [m][k]
  __shared__ bf16 Btl[2][64][32];    //  8 KB double-buffered [n][k]

  const int tid  = threadIdx.x;
  const int wave = tid >> 5;
  const int lane = tid & 31;
  const int r16  = lane & 15;
  const int hi   = lane >> 4;          // 0: lanes 0-15, 1: lanes 16-31
  const int kA   = hi ? 8 : 0;         // A-frag chunk bases: kA, kA+16
  const int kB   = hi ? 8 : 0;         // B-frag chunk bases: kB, kB+16

  // A staging: 4 float4 per thread. B staging: 1 async b128 chunk per thread.
  const int ar[4] = { (tid + 0)   >> 3, (tid + 256) >> 3,
                      (tid + 512) >> 3, (tid + 768) >> 3 };
  const int ac    = (tid & 7) * 4;
  const int bn    = tid >> 2;          // B row (n) 0..63
  const int bk8   = (tid & 3) * 8;     // B k-chunk 0/8/16/24

  const floatx8 vzero = {};
  floatx8 acc[4];
  for (int i = 0; i < 4; ++i) acc[i] = vzero;

  // ---- prologue: fill buffer 0 with kb = 0 ----
  async_cp16(&Wt[(size_t)(n0 + bn) * DIM + bk8], &Btl[0][bn][bk8]);
  float4 av[4];
  #pragma unroll
  for (int i = 0; i < 4; ++i)
    av[i] = *(const float4*)&src[(size_t)(m0 + ar[i]) * DIM + ac];
  #pragma unroll
  for (int i = 0; i < 4; ++i) *(bf16x4*)&Asl[0][ar[i]][ac] = cvt4(av[i]);
  wait_async();
  __syncthreads();

  for (int it = 0; it < DIM / 32; ++it) {
    const int cur = it & 1, nxt = cur ^ 1;
    const int kbn = (it < DIM / 32 - 1) ? (it + 1) * 32 : 0;   // branchless tail

    // issue next-step staging (lands during WMMA below)
    async_cp16(&Wt[(size_t)(n0 + bn) * DIM + kbn + bk8], &Btl[nxt][bn][bk8]);
    __builtin_prefetch(&src[(size_t)(m0 + ar[0]) * DIM + kbn + ac], 0, 3);
    #pragma unroll
    for (int i = 0; i < 4; ++i)
      av[i] = *(const float4*)&src[(size_t)(m0 + ar[i]) * DIM + kbn + ac];

    // compute from current buffers
    const bf16* arow = &Asl[cur][wave * 16 + r16][0];
    bf16x16 afrag = load_frag(arow + kA, arow + kA + 16);
    bf16x16 bfrag[4];
    #pragma unroll
    for (int ns = 0; ns < 4; ++ns) {
      const bf16* brow = &Btl[cur][ns * 16 + r16][kB];
      bfrag[ns] = load_frag(brow, brow + 16);
    }
    #pragma unroll
    for (int ns = 0; ns < 4; ++ns) acc[ns] = wmma_bf16(afrag, bfrag[ns], acc[ns]);

    // stage A into next buffer, then fence async B + barrier
    #pragma unroll
    for (int i = 0; i < 4; ++i) *(bf16x4*)&Asl[nxt][ar[i]][ac] = cvt4(av[i]);
    wait_async();
    __syncthreads();
  }

  #pragma unroll
  for (int ns = 0; ns < 4; ++ns) {
    const int gc = n0 + ns * 16 + r16;       // output column (0..767)
    const int hh = gc >> 6, dd = gc & 63;    // head, depth
    const float bval = bias[gc];
    #pragma unroll
    for (int vv = 0; vv < 8; ++vv) {
      const int gr = m0 + wave * 16 + vv + hi * 8;   // output row (0..8191)
      const int bb = gr >> 9, ss = gr & 511;
      const float val = acc[ns][vv] + bval;
      if (gemm < 2)
        dst[(((size_t)(bb * NH + hh)) * SEQ + ss) * DHD + dd] = (bf16)val;
      else
        dst[(((size_t)(bb * NH + hh)) * DHD + dd) * SEQ + ss] = (bf16)val;
    }
  }
}

// ---------------------------------------------------------------------------
// Kernel 2: attention. grid = B*H*(S/16) = 6144 blocks, 256 threads.
// Per block: 16 query rows of one (b,h). Logits LDS-resident (fp32 32 KB),
// bias fused into coalesced softmax pass, ctx = P x V via 8-way split-K WMMA.
// ---------------------------------------------------------------------------
__global__ void __launch_bounds__(256) attention_kernel(
    const float* __restrict__ mask, const float* __restrict__ adjoin,
    const bf16* __restrict__ Qh, const bf16* __restrict__ Kh,
    const bf16* __restrict__ VhT,
    float* __restrict__ attn_out, bf16* __restrict__ ctx)
{
  const int bh = blockIdx.x >> 5;        // (b*NH + h)
  const int qb = blockIdx.x & 31;
  const int b  = bh / NH;
  const int h  = bh % NH;
  const int q0 = qb * 16;

  const bf16* qp = Qh  + (size_t)bh * SEQ * DHD;
  const bf16* kp = Kh  + (size_t)bh * SEQ * DHD;
  const bf16* vp = VhT + (size_t)bh * DHD * SEQ;

  __shared__ float logits[16][SEQ];      // 32 KB (reused as part[8][16][64])
  __shared__ bf16  ptile[16][SEQ];       // 16 KB
  __shared__ bf16  qtile[16][DHD];       //  2 KB
  __shared__ float rowmax[16];
  __shared__ float rowinv[16];

  const int tid  = threadIdx.x;
  const int wave = tid >> 5;
  const int lane = tid & 31;
  const int r16  = lane & 15;
  const int hi   = lane >> 4;
  const int kA   = hi ? 8 : 0;
  const int kB   = hi ? 8 : 0;

  {                                      // stage Q tile 16x64 (bf16x4 copies)
    const int r = tid >> 4, c = (tid & 15) * 4;
    *(bf16x4*)&qtile[r][c] = *(const bf16x4*)&qp[(size_t)(q0 + r) * DHD + c];
  }
  __syncthreads();

  // ---- raw scores = (Q K^T)/8 ; wave w -> keys [w*64, w*64+64) ----
  {
    const bf16* arow = &qtile[r16][0];
    bf16x16 a0 = load_frag(arow + kA,      arow + kA + 16);       // K 0..31
    bf16x16 a1 = load_frag(arow + 32 + kA, arow + 32 + kA + 16);  // K 32..63
    bf16x16 b0[4], b1[4];
    #pragma unroll
    for (int ns = 0; ns < 4; ++ns) {
      const bf16* krow = kp + (size_t)(wave * 64 + ns * 16 + r16) * DHD + kB;
      b0[ns] = load_frag(krow,      krow + 16);
      b1[ns] = load_frag(krow + 32, krow + 48);
    }
    #pragma unroll
    for (int ns = 0; ns < 4; ++ns) {
      floatx8 acc = {};
      acc = wmma_bf16(a0, b0[ns], acc);
      acc = wmma_bf16(a1, b1[ns], acc);
      const int gc = wave * 64 + ns * 16 + r16;
      #pragma unroll
      for (int vv = 0; vv < 8; ++vv) logits[vv + hi * 8][gc] = acc[vv] * 0.125f;
    }
  }
  __syncthreads();

  // ---- bias + softmax stats: 16 threads/row, float4, coalesced mask/adjoin ----
  {
    const int r = tid >> 4, li = tid & 15;
    const float* maskrow = mask   + ((size_t)b * SEQ + q0 + r) * SEQ;
    const float* adjrow  = adjoin + ((size_t)b * SEQ + q0 + r) * SEQ;
    float m = -3.4e38f;
    #pragma unroll
    for (int i = 0; i < 8; ++i) {
      const int c = (li + 16 * i) * 4;
      float4 x  = *(const float4*)&logits[r][c];
      float4 mk = *(const float4*)&maskrow[c];
      float4 ad = *(const float4*)&adjrow[c];
      x.x += mk.x * (-1e9f) + ad.x;
      x.y += mk.y * (-1e9f) + ad.y;
      x.z += mk.z * (-1e9f) + ad.z;
      x.w += mk.w * (-1e9f) + ad.w;
      *(float4*)&logits[r][c] = x;
      m = fmaxf(m, fmaxf(fmaxf(x.x, x.y), fmaxf(x.z, x.w)));
    }
    #pragma unroll
    for (int off = 8; off > 0; off >>= 1) m = fmaxf(m, __shfl_xor(m, off, 32));
    float s = 0.f;
    #pragma unroll
    for (int i = 0; i < 8; ++i) {
      const int c = (li + 16 * i) * 4;
      float4 x = *(const float4*)&logits[r][c];
      s += __expf(x.x - m) + __expf(x.y - m) + __expf(x.z - m) + __expf(x.w - m);
    }
    #pragma unroll
    for (int off = 8; off > 0; off >>= 1) s += __shfl_xor(s, off, 32);
    if (li == 0) { rowmax[r] = m; rowinv[r] = 1.0f / s; }
  }
  __syncthreads();

  // ---- P = softmax: bf16x4 to LDS + float4 to global attn (coalesced) ----
  #pragma unroll
  for (int i = 0; i < 8; ++i) {
    const int idx = tid + 256 * i;           // 2048 float4 groups
    const int r = idx >> 7, c = (idx & 127) * 4;
    const float m = rowmax[r], inv = rowinv[r];
    float4 x = *(const float4*)&logits[r][c];
    float4 p;
    p.x = __expf(x.x - m) * inv;
    p.y = __expf(x.y - m) * inv;
    p.z = __expf(x.z - m) * inv;
    p.w = __expf(x.w - m) * inv;
    bf16x4 pb = { (bf16)p.x, (bf16)p.y, (bf16)p.z, (bf16)p.w };
    *(bf16x4*)&ptile[r][c] = pb;
    *(float4*)&attn_out[((size_t)bh * SEQ + q0 + r) * SEQ + c] = p;
  }
  __syncthreads();   // everyone done reading logits; safe to reuse as partials

  // ---- ctx: split-K across waves, wave w -> keys [w*64, w*64+64) ----
  float* part = &logits[0][0];           // [8][16][64] fp32 partials
  {
    const int kb0 = wave * 64;
    const bf16* arow = &ptile[r16][kb0];
    bf16x16 a0 = load_frag(arow + kA,      arow + kA + 16);
    bf16x16 a1 = load_frag(arow + 32 + kA, arow + 32 + kA + 16);
    bf16x16 b0[4], b1[4];
    #pragma unroll
    for (int ns = 0; ns < 4; ++ns) {
      const bf16* vrow = vp + (size_t)(ns * 16 + r16) * SEQ + kb0 + kB;
      b0[ns] = load_frag(vrow,      vrow + 16);
      b1[ns] = load_frag(vrow + 32, vrow + 48);
    }
    floatx8 acc[4];
    #pragma unroll
    for (int ns = 0; ns < 4; ++ns) {
      acc[ns] = (floatx8){};
      acc[ns] = wmma_bf16(a0, b0[ns], acc[ns]);
      acc[ns] = wmma_bf16(a1, b1[ns], acc[ns]);
    }
    #pragma unroll
    for (int ns = 0; ns < 4; ++ns)
      #pragma unroll
      for (int vv = 0; vv < 8; ++vv)
        part[(wave * 16 + vv + hi * 8) * 64 + ns * 16 + r16] = acc[ns][vv];
  }
  __syncthreads();

  // ---- reduce split-K partials (b128 reads), store ctx bf16 (B,S,D) ----
  {
    const int r = tid >> 4, c = (tid & 15) * 4;   // 256 float4 groups = all 1024
    float4 s = make_float4(0.f, 0.f, 0.f, 0.f);
    #pragma unroll
    for (int ww = 0; ww < 8; ++ww) {
      float4 x = *(const float4*)&part[(ww * 16 + r) * 64 + c];
      s.x += x.x; s.y += x.y; s.z += x.z; s.w += x.w;
    }
    bf16x4 t = { (bf16)s.x, (bf16)s.y, (bf16)s.z, (bf16)s.w };
    *(bf16x4*)&ctx[((size_t)b * SEQ + q0 + r) * DIM + h * DHD + c] = t;
  }
}

// ---------------------------------------------------------------------------
// Kernel 3: out = ctx @ Wo + bo (fp32 output). grid=768, block=256.
// Fully async-staged: both A (ctx bf16) and B (Wo^T bf16) tiles go
// global -> LDS with no VGPR bounce; double-buffered, one barrier/iter.
// ---------------------------------------------------------------------------
__global__ void __launch_bounds__(256) out_proj_kernel(
    const bf16* __restrict__ ctxp, const bf16* __restrict__ WoT,
    const float* __restrict__ bo, float* __restrict__ out)
{
  const int tile = blockIdx.x;
  const int m0 = (tile / (DIM / 64)) * 128;
  const int n0 = (tile % (DIM / 64)) * 64;

  __shared__ bf16 Asl[2][128][32];
  __shared__ bf16 Btl[2][64][32];

  const int tid  = threadIdx.x;
  const int wave = tid >> 5;
  const int lane = tid & 31;
  const int r16  = lane & 15;
  const int hi   = lane >> 4;
  const int kA   = hi ? 8 : 0;
  const int kB   = hi ? 8 : 0;

  // A: 512 16B chunks -> 2 per thread; B: 256 16B chunks -> 1 per thread.
  const int am[2] = { (tid + 0) >> 2, (tid + 256) >> 2 };   // A rows
  const int ak8   = (tid & 3) * 8;
  const int bn    = tid >> 2;
  const int bk8   = (tid & 3) * 8;

  const floatx8 vzero = {};
  floatx8 acc[4];
  for (int i = 0; i < 4; ++i) acc[i] = vzero;

  // ---- prologue: fill buffer 0 with kb = 0 ----
  #pragma unroll
  for (int i = 0; i < 2; ++i)
    async_cp16(&ctxp[(size_t)(m0 + am[i]) * DIM + ak8], &Asl[0][am[i]][ak8]);
  async_cp16(&WoT[(size_t)(n0 + bn) * DIM + bk8], &Btl[0][bn][bk8]);
  wait_async();
  __syncthreads();

  for (int it = 0; it < DIM / 32; ++it) {
    const int cur = it & 1, nxt = cur ^ 1;
    const int kbn = (it < DIM / 32 - 1) ? (it + 1) * 32 : 0;

    #pragma unroll
    for (int i = 0; i < 2; ++i)
      async_cp16(&ctxp[(size_t)(m0 + am[i]) * DIM + kbn + ak8],
                 &Asl[nxt][am[i]][ak8]);
    async_cp16(&WoT[(size_t)(n0 + bn) * DIM + kbn + bk8], &Btl[nxt][bn][bk8]);

    const bf16* arow = &Asl[cur][wave * 16 + r16][0];
    bf16x16 afrag = load_frag(arow + kA, arow + kA + 16);
    bf16x16 bfrag[4];
    #pragma unroll
    for (int ns = 0; ns < 4; ++ns) {
      const bf16* brow = &Btl[cur][ns * 16 + r16][kB];
      bfrag[ns] = load_frag(brow, brow + 16);
    }
    #pragma unroll
    for (int ns = 0; ns < 4; ++ns) acc[ns] = wmma_bf16(afrag, bfrag[ns], acc[ns]);

    wait_async();
    __syncthreads();
  }

  #pragma unroll
  for (int ns = 0; ns < 4; ++ns) {
    const int gc = n0 + ns * 16 + r16;
    const float bval = bo[gc];
    #pragma unroll
    for (int vv = 0; vv < 8; ++vv) {
      const int gr = m0 + wave * 16 + vv + hi * 8;
      out[(size_t)gr * DIM + gc] = acc[ns][vv] + bval;
    }
  }
}

// ---------------------------------------------------------------------------
extern "C" void kernel_launch(void* const* d_in, const int* in_sizes, int n_in,
                              void* d_out, int out_size, void* d_ws, size_t ws_size,
                              hipStream_t stream) {
  const float* q      = (const float*)d_in[0];
  const float* k      = (const float*)d_in[1];
  const float* v      = (const float*)d_in[2];
  const float* mask   = (const float*)d_in[3];
  const float* adjoin = (const float*)d_in[4];
  const float* Wq = (const float*)d_in[5];
  const float* bq = (const float*)d_in[6];
  const float* Wk = (const float*)d_in[7];
  const float* bk = (const float*)d_in[8];
  const float* Wv = (const float*)d_in[9];
  const float* bv = (const float*)d_in[10];
  const float* Wo = (const float*)d_in[11];
  const float* bo = (const float*)d_in[12];

  float* out  = (float*)d_out;
  float* attn = out + (size_t)BATCH * SEQ * DIM;   // tuple: (out, attn)

  const size_t EH = (size_t)BATCH * SEQ * DIM;     // 6,291,456 elements
  const size_t WE = (size_t)DIM * DIM;             //   589,824 elements
  bf16* Qh  = (bf16*)d_ws;
  bf16* Kh  = Qh + EH;
  bf16* VhT = Qh + 2 * EH;
  bf16* ctx = Qh + 3 * EH;
  bf16* WT  = Qh + 4 * EH;                         // 4 transposed weights (~55 MB total)

  const int gemm_tiles = (DIM / 64) * ((BATCH * SEQ) / 128);   // 12*64 = 768

  prep_wt_kernel<<<dim3((DIM / 64) * (DIM / 64), 4), 256, 0, stream>>>(
      Wq, Wk, Wv, Wo, WT);

  qkv_proj_kernel<<<dim3(gemm_tiles, 3), 256, 0, stream>>>(
      q, k, v, WT, bq, bk, bv, Qh, Kh, VhT);

  attention_kernel<<<BATCH * NH * (SEQ / 16), 256, 0, stream>>>(
      mask, adjoin, Qh, Kh, VhT, attn, ctx);

  out_proj_kernel<<<gemm_tiles, 256, 0, stream>>>(ctx, WT + 3 * WE, bo, out);
}
